// ProtoHead_37297495999027
// MI455X (gfx1250) — matrix operand order repaired and checked by reading, not compiled, not run on hardware
//
#include <hip/hip_runtime.h>
#include <hip/hip_bf16.h>

// ---------------------------------------------------------------------------
// ProtoNet head for MI455X (gfx1250, wave32, WMMA).
// logits[m][n] = -( ||q_m||^2 + ||p_n||^2 - 2 * <q_m, p_n> )
// Core GEMM (65536 x 1024 x 512, fp32 data) via split-bf16 emulated fp32:
// x = hi + lo (bf16); x*y ~= hi*hi + hi*lo + lo*hi, f32-accumulated with
// v_wmma_f32_16x16x32_bf16. Double-buffered LDS; B tiles streamed with
// global_load_async_to_lds_b128 (ASYNCcnt) overlapping the WMMA pipe.
// ---------------------------------------------------------------------------

typedef __attribute__((ext_vector_type(16))) __bf16 v16bf;
typedef __attribute__((ext_vector_type(4)))  __bf16 v4bf;
typedef __attribute__((ext_vector_type(8)))  float  v8f;

#define NS 65536
#define NQ 65536
#define DIM 512
#define NCLS 1024
#define LDS_STRIDE 40            // 32 bf16 + 8 pad = 80B rows: conflict-free, 16B aligned
#define SEG (128 * LDS_STRIDE)   // one 128-row tile segment (elements)

__device__ __forceinline__ __bf16 f32_to_bf16(float x) {
    union { float f; unsigned u; } a; a.f = x;
    unsigned r = a.u + 0x7fffu + ((a.u >> 16) & 1u);   // round-to-nearest-even
    union { unsigned short s; __bf16 b; } o;
    o.s = (unsigned short)(r >> 16);
    return o.b;
}
__device__ __forceinline__ float bf16_to_f32(__bf16 b) {
    union { unsigned short s; __bf16 b; } i; i.b = b;
    union { unsigned u; float f; } o; o.u = ((unsigned)i.s) << 16;
    return o.f;
}

// gfx1250 async global->LDS copy, 16 bytes per lane (tracked by ASYNCcnt).
__device__ __forceinline__ void async_copy_b128(void* lds_ptr, const void* gptr) {
    unsigned lds_off = (unsigned)(unsigned long long)lds_ptr;   // LDS offset = low 32b
    unsigned long long ga = (unsigned long long)gptr;
    asm volatile("global_load_async_to_lds_b128 %0, %1, off"
                 :: "v"(lds_off), "v"(ga) : "memory");
}
__device__ __forceinline__ void wait_asynccnt0() {
    asm volatile("s_wait_asynccnt 0x0" ::: "memory");
}

// ---------------- Phase 1: scatter-add support vectors into class sums ------
__global__ __launch_bounds__(256) void proto_scatter_kernel(
    const float* __restrict__ zs, const int* __restrict__ ys,
    float* __restrict__ sums, float* __restrict__ counts) {
    int gid = blockIdx.x * 256 + threadIdx.x;       // one float4 per thread
    int row = gid >> 7;                             // 128 float4-chunks per row
    int d4  = (gid & 127) << 2;
    int c = ys[row];
    const float4 v = *(const float4*)(zs + (size_t)row * DIM + d4);
    float* s = sums + (size_t)c * DIM + d4;
    atomicAdd(s + 0, v.x);
    atomicAdd(s + 1, v.y);
    atomicAdd(s + 2, v.z);
    atomicAdd(s + 3, v.w);
    if (d4 == 0) atomicAdd(counts + c, 1.0f);
}

// ---------------- Phase 2: protos = sums/count, bf16 hi/lo split, ||p||^2 ---
__global__ __launch_bounds__(256) void proto_finalize_kernel(
    const float* __restrict__ sums, const float* __restrict__ counts,
    __bf16* __restrict__ phi, __bf16* __restrict__ plo,
    float* __restrict__ psq) {
    __shared__ float red[256];
    int c = blockIdx.x;
    int t = threadIdx.x;
    float cnt = counts[c];
    float inv = 1.0f / fmaxf(cnt, 1.0f);
    float local = 0.0f;
    for (int i = t; i < DIM; i += 256) {
        float mean = sums[(size_t)c * DIM + i] * inv;
        __bf16 hi = f32_to_bf16(mean);
        float hif = bf16_to_f32(hi);
        __bf16 lo = f32_to_bf16(mean - hif);
        phi[(size_t)c * DIM + i] = hi;
        plo[(size_t)c * DIM + i] = lo;
        local += mean * mean;
    }
    red[t] = local;
    __syncthreads();
    for (int s = 128; s > 0; s >>= 1) {
        if (t < s) red[t] += red[t + s];
        __syncthreads();
    }
    if (t == 0) psq[c] = red[0];
}

// ---------------- Phase 3: ||q||^2 per query row (one wave32 per row) -------
__global__ __launch_bounds__(256) void qsq_kernel(
    const float* __restrict__ zq, float* __restrict__ qsq) {
    int lane = threadIdx.x & 31;
    int wave = threadIdx.x >> 5;
    int row = blockIdx.x * 8 + wave;
    const float4* p = (const float4*)(zq + (size_t)row * DIM);
    float s = 0.0f;
#pragma unroll
    for (int j = 0; j < 4; ++j) {
        float4 f = p[lane + j * 32];
        s += f.x * f.x + f.y * f.y + f.z * f.z + f.w * f.w;
    }
#pragma unroll
    for (int off = 16; off > 0; off >>= 1)
        s += __shfl_xor(s, off, 32);
    if (lane == 0) qsq[row] = s;
}

// ---------------- Phase 4: fused split-bf16 WMMA GEMM + distance epilogue ---
// Block tile 128(M) x 128(N), BK=32, double-buffered LDS. 8 waves:
// wave = (wm 0..3) x (wn 0..1); each wave owns a 32x64 output tile (2x4 accs).
__global__ __launch_bounds__(256) void proto_gemm_kernel(
    const float* __restrict__ zq,
    const __bf16* __restrict__ phi, const __bf16* __restrict__ plo,
    const float* __restrict__ qsq, const float* __restrict__ psq,
    float* __restrict__ out) {

    // [buf][ {Ahi, Alo, Bhi, Blo} x SEG ]
    __shared__ __bf16 lds[2][4 * SEG];

    const int tid  = threadIdx.x;
    const int lane = tid & 31;
    const int wave = tid >> 5;
    const int wm = wave & 3;        // M sub-tile (32 rows each)
    const int wn = wave >> 2;       // N sub-tile (64 cols each)
    const int l15 = lane & 15;
    const int lhi = lane >> 4;

    const int m0 = blockIdx.y * 128;
    const int n0 = blockIdx.x * 128;

    // B-tile loader mapping: 2 threads per proto row, 16 bf16 (32B) each
    const int brow = tid >> 1;
    const int bk   = (tid & 1) << 4;
    const size_t bgoff = (size_t)(n0 + brow) * DIM + bk;

    v8f acc[2][4];
#pragma unroll
    for (int i = 0; i < 2; ++i)
#pragma unroll
        for (int j = 0; j < 4; ++j)
            acc[i][j] = (v8f){0.f, 0.f, 0.f, 0.f, 0.f, 0.f, 0.f, 0.f};

    // ---- tile loaders ------------------------------------------------------
    auto load_tile = [&](int k0, __bf16* buf) {
        __bf16* Ahi = buf;
        __bf16* Alo = buf + SEG;
        __bf16* Bhi = buf + 2 * SEG;
        __bf16* Blo = buf + 3 * SEG;
        // B: async DMA straight into LDS (4 x b128 per thread, ASYNCcnt)
        async_copy_b128(&Bhi[brow * LDS_STRIDE + bk],     phi + bgoff + k0);
        async_copy_b128(&Bhi[brow * LDS_STRIDE + bk + 8], phi + bgoff + k0 + 8);
        async_copy_b128(&Blo[brow * LDS_STRIDE + bk],     plo + bgoff + k0);
        async_copy_b128(&Blo[brow * LDS_STRIDE + bk + 8], plo + bgoff + k0 + 8);
        // A: 128 x 32 fp32 -> hi/lo bf16 split in registers -> LDS
#pragma unroll
        for (int it = 0; it < 4; ++it) {
            int q   = tid + it * 256;
            int row = q >> 3;
            int c4  = (q & 7) << 2;
            float4 v = *(const float4*)(zq + (size_t)(m0 + row) * DIM + k0 + c4);
            __bf16 h0 = f32_to_bf16(v.x), h1 = f32_to_bf16(v.y);
            __bf16 h2 = f32_to_bf16(v.z), h3 = f32_to_bf16(v.w);
            __bf16 g0 = f32_to_bf16(v.x - bf16_to_f32(h0));
            __bf16 g1 = f32_to_bf16(v.y - bf16_to_f32(h1));
            __bf16 g2 = f32_to_bf16(v.z - bf16_to_f32(h2));
            __bf16 g3 = f32_to_bf16(v.w - bf16_to_f32(h3));
            *(v4bf*)&Ahi[row * LDS_STRIDE + c4] = (v4bf){h0, h1, h2, h3};
            *(v4bf*)&Alo[row * LDS_STRIDE + c4] = (v4bf){g0, g1, g2, g3};
        }
    };

    // ---- prologue: fill buffer 0 ------------------------------------------
    load_tile(0, lds[0]);
    wait_asynccnt0();
    __syncthreads();

    int cur = 0;
    for (int k0 = 0; k0 < DIM; k0 += 32) {
        const int nxt = cur ^ 1;
        if (k0 + 32 < DIM)
            load_tile(k0 + 32, lds[nxt]);   // overlaps with WMMAs below

        const __bf16* Ahi = lds[cur];
        const __bf16* Alo = lds[cur] + SEG;
        const __bf16* Bhi = lds[cur] + 2 * SEG;
        const __bf16* Blo = lds[cur] + 3 * SEG;

        // fragments (16-bit A 16x32 layout; B mirrors via N-major rows)
        v16bf ahi[2], alo[2], bhi[4], blo[4];
        const int kbA = lhi * 8;     // lanes 16-31 hold K+8 halves
        const int kbB = lhi * 8;     // lanes 16-31 hold K+8 halves
#pragma unroll
        for (int tm = 0; tm < 2; ++tm) {
            int arow = wm * 32 + tm * 16 + l15;
            union { uint4 u[2]; v16bf v; } fh, fl;
            fh.u[0] = *(const uint4*)&Ahi[arow * LDS_STRIDE + kbA];
            fh.u[1] = *(const uint4*)&Ahi[arow * LDS_STRIDE + 16 + kbA];
            fl.u[0] = *(const uint4*)&Alo[arow * LDS_STRIDE + kbA];
            fl.u[1] = *(const uint4*)&Alo[arow * LDS_STRIDE + 16 + kbA];
            ahi[tm] = fh.v; alo[tm] = fl.v;
        }
#pragma unroll
        for (int tn = 0; tn < 4; ++tn) {
            int bcol = wn * 64 + tn * 16 + l15;
            union { uint4 u[2]; v16bf v; } fh, fl;
            fh.u[0] = *(const uint4*)&Bhi[bcol * LDS_STRIDE + kbB];
            fh.u[1] = *(const uint4*)&Bhi[bcol * LDS_STRIDE + kbB + 16];
            fl.u[0] = *(const uint4*)&Blo[bcol * LDS_STRIDE + kbB];
            fl.u[1] = *(const uint4*)&Blo[bcol * LDS_STRIDE + kbB + 16];
            bhi[tn] = fh.v; blo[tn] = fl.v;
        }

        // 3-pass split-bf16 accumulate: hi*hi + hi*lo + lo*hi
#pragma unroll
        for (int tm = 0; tm < 2; ++tm) {
#pragma unroll
            for (int tn = 0; tn < 4; ++tn) {
                acc[tm][tn] = __builtin_amdgcn_wmma_f32_16x16x32_bf16(
                    false, ahi[tm], false, bhi[tn], (short)0, acc[tm][tn], false, false);
                acc[tm][tn] = __builtin_amdgcn_wmma_f32_16x16x32_bf16(
                    false, ahi[tm], false, blo[tn], (short)0, acc[tm][tn], false, false);
                acc[tm][tn] = __builtin_amdgcn_wmma_f32_16x16x32_bf16(
                    false, alo[tm], false, bhi[tn], (short)0, acc[tm][tn], false, false);
            }
        }

        wait_asynccnt0();   // next-buffer B DMA landed (A ds-stores fenced by barrier)
        __syncthreads();
        cur = nxt;
    }

    // ---- epilogue: logits = -(qsq + psq - 2*cross), streamed non-temporal --
#pragma unroll
    for (int tm = 0; tm < 2; ++tm) {
        int mbase = m0 + wm * 32 + tm * 16 + lhi * 8;   // C/D layout: M = v (+8 hi half)
        float qs[8];
#pragma unroll
        for (int v = 0; v < 8; ++v) qs[v] = qsq[mbase + v];
#pragma unroll
        for (int tn = 0; tn < 4; ++tn) {
            int n = n0 + wn * 64 + tn * 16 + l15;       // C/D layout: N = lane&15
            float pn = psq[n];
#pragma unroll
            for (int v = 0; v < 8; ++v) {
                float cross = acc[tm][tn][v];
                float logit = -(qs[v] + pn - 2.0f * cross);
                __builtin_nontemporal_store(logit, &out[(size_t)(mbase + v) * NCLS + n]);
            }
        }
    }
}

// ---------------------------------------------------------------------------
extern "C" void kernel_launch(void* const* d_in, const int* in_sizes, int n_in,
                              void* d_out, int out_size, void* d_ws, size_t ws_size,
                              hipStream_t stream) {
    const float* z_support = (const float*)d_in[0];
    const int*   y_support = (const int*)d_in[1];
    const float* z_query   = (const float*)d_in[2];
    float* out = (float*)d_out;

    // workspace layout
    float* sums   = (float*)d_ws;                   // NCLS*DIM
    float* counts = sums + (size_t)NCLS * DIM;      // NCLS
    float* psq    = counts + NCLS;                  // NCLS
    float* qsq    = psq + NCLS;                     // NQ
    __bf16* phi   = (__bf16*)(qsq + NQ);            // NCLS*DIM
    __bf16* plo   = phi + (size_t)NCLS * DIM;       // NCLS*DIM

    hipMemsetAsync(sums, 0, (size_t)(NCLS * DIM + NCLS) * sizeof(float), stream);

    proto_scatter_kernel<<<(NS * (DIM / 4)) / 256, 256, 0, stream>>>(
        z_support, y_support, sums, counts);

    proto_finalize_kernel<<<NCLS, 256, 0, stream>>>(sums, counts, phi, plo, psq);

    qsq_kernel<<<NQ / 8, 256, 0, stream>>>(z_query, qsq);

    dim3 grid(NCLS / 128, NQ / 128);   // ntile fastest -> A-tile reuse in L2
    proto_gemm_kernel<<<grid, 256, 0, stream>>>(z_query, phi, plo, qsq, psq, out);
}
